// attention_module_59313498357813
// MI455X (gfx1250) — matrix operand order repaired and checked by reading, not compiled, not run on hardware
//
#include <hip/hip_runtime.h>
#include <hip/hip_bf16.h>
#include <cstddef>
#include <cstdint>

typedef __attribute__((ext_vector_type(16))) _Float16 v16h;
typedef __attribute__((ext_vector_type(8)))  float    v8f;

#define CC   256   // channels
#define DD   128   // attention dim
#define HWSZ 4096  // H*W
#define NB   4     // batch
#define LDP  8     // LDS row padding (halves)

// ---- CDNA5 async global->LDS copy (ASYNCcnt path, ISA ch.10.7/15.18) ------
__device__ __forceinline__ void async_ld16(uint32_t lds_off, const void* gptr) {
    asm volatile("global_load_async_to_lds_b128 %0, %1, off"
                 :: "v"(lds_off), "v"((uint64_t)(uintptr_t)gptr)
                 : "memory");
}

// ---- WMMA fragment helpers (CDNA5 ISA 7.12.2 layouts, wave32) ------------

// A fragment 16x32 f16 from an LDS tile stored row-major [row][k], row stride ld.
__device__ __forceinline__ v16h load_a_frag(const _Float16* src, int row0, int k0, int ld) {
    const int lane = threadIdx.x & 31;
    const int m = lane & 15, hi = lane >> 4;
    const _Float16* p = src + (row0 + m) * ld + k0 + hi * 8;
    v16h r;
#pragma unroll
    for (int i = 0; i < 8; ++i) r[i] = p[i];
#pragma unroll
    for (int i = 0; i < 8; ++i) r[8 + i] = p[16 + i];
    return r;
}

// B fragment 32x16 f16. Source stored row-major per COLUMN entity: src[n][k].
__device__ __forceinline__ v16h load_b_frag(const _Float16* src, int n0, int k0, int ld) {
    const int lane = threadIdx.x & 31;
    const int n = lane & 15, hi = lane >> 4;
    const _Float16* p = src + (n0 + n) * ld + k0 + hi * 8;
    v16h r;
#pragma unroll
    for (int i = 0; i < 8; ++i) r[i] = p[i];
#pragma unroll
    for (int i = 0; i < 8; ++i) r[8 + i] = p[16 + i];
    return r;
}

__device__ __forceinline__ v8f wmma_f16(v16h a, v16h b, v8f c) {
    return __builtin_amdgcn_wmma_f32_16x16x32_f16(false, a, false, b, (short)0, c, false, false);
}

// ---- Kernel 1: g/f/h = W @ x  (per batch GEMM [HW x d] = [HW x C][C x d]) --
__global__ __launch_bounds__(256) void proj_kernel(
    const float* __restrict__ x,                       // [B, C, HW]
    const float* __restrict__ Wg, const float* __restrict__ Wf, const float* __restrict__ Wh,
    _Float16* __restrict__ g, _Float16* __restrict__ f, _Float16* __restrict__ h) {
    __shared__ _Float16 xs[128 * (CC + LDP)];          // [token][c]
    __shared__ _Float16 ws[DD  * (CC + LDP)];          // [dout][c]
    const int tid  = threadIdx.x;
    const int wave = tid >> 5;
    const int b    = blockIdx.x >> 5;
    const int t0   = (blockIdx.x & 31) * 128;
    const int which = blockIdx.y;
    const float*  W   = (which == 0) ? Wg : (which == 1) ? Wf : Wh;
    _Float16*     out = (which == 0) ? g  : (which == 1) ? f  : h;

    {
        const int t = tid & 127;
        for (int c = tid >> 7; c < CC; c += 2)
            xs[t * (CC + LDP) + c] = (_Float16)x[((size_t)b * CC + c) * HWSZ + t0 + t];
    }
    for (int idx = tid; idx < DD * CC; idx += 256) {
        const int d = idx >> 8, c = idx & 255;
        ws[d * (CC + LDP) + c] = (_Float16)W[idx];
    }
    __syncthreads();

    v8f acc[8] = {};
#pragma unroll 1
    for (int kc = 0; kc < CC; kc += 32) {
        const v16h a = load_a_frag(xs, wave * 16, kc, CC + LDP);
#pragma unroll
        for (int j = 0; j < 8; ++j) {
            const v16h bf = load_b_frag(ws, j * 16, kc, CC + LDP);
            acc[j] = wmma_f16(a, bf, acc[j]);
        }
    }
    const int lane = tid & 31, nn = lane & 15, hi = lane >> 4;
#pragma unroll
    for (int j = 0; j < 8; ++j)
#pragma unroll
        for (int r = 0; r < 8; ++r) {
            const int tok = t0 + wave * 16 + r + 8 * hi;
            out[((size_t)b * HWSZ + tok) * DD + j * 16 + nn] = (_Float16)acc[j][r];
        }
}

// ---- Kernel 2: flash attention with async double-buffered K/V staging -----
// grid.x = NB * (HWSZ/128); block = 256 (8 waves, 16 query rows per wave)
__global__ __launch_bounds__(256) void attn_kernel(
    const _Float16* __restrict__ g, const _Float16* __restrict__ f,
    const _Float16* __restrict__ h, _Float16* __restrict__ vout) {
    const int LD = DD + LDP;                            // 136
    __shared__ _Float16 qs [128 * (DD + LDP)];          // [q][d]
    __shared__ _Float16 hs2[2][128 * (DD + LDP)];       // [key][d] double buffer (async dest)
    __shared__ _Float16 fsr[2][128 * DD];               // raw f tiles [key][d] (async dest)
    __shared__ _Float16 fst[DD * (DD + LDP)];           // [d][key] for P@V B-fragments
    __shared__ _Float16 pws[8 * 16 * (DD + LDP)];       // per-wave P staging
    const int tid = threadIdx.x, wave = tid >> 5, lane = tid & 31;
    const int b  = blockIdx.x >> 5;
    const int q0 = (blockIdx.x & 31) * 128;
    const int NKT = HWSZ / 128;

    // issue one key tile's async copies: 8x16B per tensor per thread (16 total)
    auto issue_tile = [&](int kt, int buf) {
        const size_t base = ((size_t)b * HWSZ + (size_t)kt * 128) * DD;
#pragma unroll
        for (int c = 0; c < 8; ++c) {
            const int e = (tid + c * 256) * 8;          // element index in 128x128 tile
            const int k = e >> 7, d = e & 127;          // d is a multiple of 8 -> 16B aligned
            async_ld16((uint32_t)(uintptr_t)&hs2[buf][k * LD + d], h + base + k * DD + d);
            async_ld16((uint32_t)(uintptr_t)&fsr[buf][k * DD + d], f + base + k * DD + d);
        }
    };

    issue_tile(0, 0);

    for (int idx = tid; idx < 128 * DD; idx += 256) {
        const int q = idx >> 7, d = idx & 127;
        qs[q * LD + d] = g[((size_t)b * HWSZ + q0 + q) * DD + d];
    }

    v8f   o[8] = {};
    float mrow[8], lrow[8];
#pragma unroll
    for (int r = 0; r < 8; ++r) { mrow[r] = -1e30f; lrow[r] = 0.0f; }

    _Float16* pw = pws + wave * 16 * LD;
    const int nn = lane & 15, hi = lane >> 4;

    for (int kt = 0; kt < NKT; ++kt) {
        const int cur = kt & 1;
        if (kt + 1 < NKT) {
            issue_tile(kt + 1, cur ^ 1);
            // <=16 outstanding => tile kt has fully landed; kt+1 still in flight
            asm volatile("s_wait_asynccnt 16" ::: "memory");
        } else {
            asm volatile("s_wait_asynccnt 0" ::: "memory");
        }
        __syncthreads();                                // barrier A: tile kt visible

        // S = Q @ K^T : 16 queries x 128 keys
        v8f s[8] = {};
#pragma unroll 1
        for (int kc = 0; kc < DD; kc += 32) {
            const v16h a = load_a_frag(qs, wave * 16, kc, LD);
#pragma unroll
            for (int j = 0; j < 8; ++j) {
                const v16h bf = load_b_frag(hs2[cur], j * 16, kc, LD);
                s[j] = wmma_f16(a, bf, s[j]);
            }
        }

        // cooperative LDS->LDS transpose of the V tile (overlapped with softmax)
        for (int idx = tid; idx < 128 * DD; idx += 256) {
            const int k = idx >> 7, d = idx & 127;
            fst[d * LD + k] = fsr[cur][k * DD + d];
        }

        // online softmax: lane holds 8 rows (r+8*hi), one column per subtile
#pragma unroll
        for (int r = 0; r < 8; ++r) {
            float mx = s[0][r];
#pragma unroll
            for (int j = 1; j < 8; ++j) mx = fmaxf(mx, s[j][r]);
#pragma unroll
            for (int off = 1; off < 16; off <<= 1) mx = fmaxf(mx, __shfl_xor(mx, off, 32));
            const float mnew  = fmaxf(mrow[r], mx);
            const float scale = __expf(mrow[r] - mnew);
            float sum = 0.0f;
#pragma unroll
            for (int j = 0; j < 8; ++j) {
                const float p = __expf(s[j][r] - mnew);
                s[j][r] = p;
                sum += p;
            }
#pragma unroll
            for (int off = 1; off < 16; off <<= 1) sum += __shfl_xor(sum, off, 32);
            lrow[r] = lrow[r] * scale + sum;
            mrow[r] = mnew;
#pragma unroll
            for (int nd = 0; nd < 8; ++nd) o[nd][r] *= scale;
        }

        // stage P (f16) to per-wave LDS strip: C/D layout -> row-major [16][128]
#pragma unroll
        for (int j = 0; j < 8; ++j)
#pragma unroll
            for (int r = 0; r < 8; ++r)
                pw[(r + 8 * hi) * LD + j * 16 + nn] = (_Float16)s[j][r];

        __syncthreads();                                // barrier B: fst complete

        // O += P @ V : A = P [16 x 128keys], B = F^T stored [d][key]
#pragma unroll 1
        for (int kc = 0; kc < 128; kc += 32) {
            const v16h a = load_a_frag(pw, 0, kc, LD);
#pragma unroll
            for (int nd = 0; nd < 8; ++nd) {
                const v16h bf = load_b_frag(fst, nd * 16, kc, LD);
                o[nd] = wmma_f16(a, bf, o[nd]);
            }
        }
    }

    // normalize and emit v (f16, [B, HW, d])
#pragma unroll
    for (int r = 0; r < 8; ++r) {
        const float inv = 1.0f / lrow[r];
        const int q = q0 + wave * 16 + r + 8 * hi;
#pragma unroll
        for (int nd = 0; nd < 8; ++nd)
            vout[((size_t)b * HWSZ + q) * DD + nd * 16 + nn] = (_Float16)(o[nd][r] * inv);
    }
}

// ---- Kernel 3: out = x + lambda * (Wv @ v) --------------------------------
__global__ __launch_bounds__(256) void outproj_kernel(
    const float* __restrict__ x, const float* __restrict__ Wv,
    const _Float16* __restrict__ v, const float* __restrict__ lambda_att,
    float* __restrict__ out) {
    const int LD = DD + LDP;
    __shared__ _Float16 wvs[CC * (DD + LDP)];           // [c][d] == A row-major
    __shared__ _Float16 vs [64 * (DD + LDP)];           // [token][d] == B [n][k]
    const int tid = threadIdx.x, wave = tid >> 5, lane = tid & 31;
    const int b  = blockIdx.x >> 6;
    const int t0 = (blockIdx.x & 63) * 64;
    const float lam = lambda_att[0];

    for (int idx = tid; idx < CC * DD; idx += 256) {
        const int c = idx >> 7, d = idx & 127;
        wvs[c * LD + d] = (_Float16)Wv[idx];
    }
    for (int idx = tid; idx < 64 * DD; idx += 256) {
        const int t = idx >> 7, d = idx & 127;
        vs[t * LD + d] = v[((size_t)b * HWSZ + t0 + t) * DD + d];
    }
    __syncthreads();

    v8f acc[2][4] = {};
#pragma unroll 1
    for (int kc = 0; kc < DD; kc += 32) {
        const v16h a0 = load_a_frag(wvs, wave * 32,      kc, LD);
        const v16h a1 = load_a_frag(wvs, wave * 32 + 16, kc, LD);
#pragma unroll
        for (int nt = 0; nt < 4; ++nt) {
            const v16h bf = load_b_frag(vs, nt * 16, kc, LD);
            acc[0][nt] = wmma_f16(a0, bf, acc[0][nt]);
            acc[1][nt] = wmma_f16(a1, bf, acc[1][nt]);
        }
    }
    const int nn = lane & 15, hi = lane >> 4;
#pragma unroll
    for (int mi = 0; mi < 2; ++mi)
#pragma unroll
        for (int nt = 0; nt < 4; ++nt)
#pragma unroll
            for (int r = 0; r < 8; ++r) {
                const int c = wave * 32 + mi * 16 + r + 8 * hi;
                const int t = t0 + nt * 16 + nn;        // lanes -> consecutive t (coalesced)
                const size_t idx = ((size_t)b * CC + c) * HWSZ + t;
                out[idx] = x[idx] + lam * acc[mi][nt][r];
            }
}

// ---- launcher --------------------------------------------------------------
extern "C" void kernel_launch(void* const* d_in, const int* in_sizes, int n_in,
                              void* d_out, int out_size, void* d_ws, size_t ws_size,
                              hipStream_t stream) {
    const float* x   = (const float*)d_in[0];
    const float* Wg  = (const float*)d_in[1];
    const float* Wf  = (const float*)d_in[2];
    const float* Wh  = (const float*)d_in[3];
    const float* Wv  = (const float*)d_in[4];
    const float* lam = (const float*)d_in[5];

    const size_t tok = (size_t)NB * HWSZ * DD;          // elements per f16 tensor
    _Float16* g = (_Float16*)d_ws;
    _Float16* f = g + tok;
    _Float16* h = f + tok;
    _Float16* v = h + tok;

    dim3 gp(NB * (HWSZ / 128), 3);
    proj_kernel<<<gp, 256, 0, stream>>>(x, Wg, Wf, Wh, g, f, h);
    attn_kernel<<<NB * (HWSZ / 128), 256, 0, stream>>>(g, f, h, v);
    outproj_kernel<<<NB * (HWSZ / 64), 256, 0, stream>>>(x, Wv, v, lam, (float*)d_out);
}
